// SequenceHead_60017872994825
// MI455X (gfx1250) — compile-verified, matrix operand order checked by reading, not run on hardware
//
#include <hip/hip_runtime.h>

// ---------------------------------------------------------------------------
// SequenceHead fused kernel for gfx1250 (MI455X), bf16 WMMA pipeline, v3.
// B=4, N=16384, CT=384, ATOMS=14. 65536 tokens; 256 tokens/block;
// 8 waves x 2 sequential 16-token tiles per wave.
// v3: nested weight staging (no div/mod), 2 tiles/wave, global prefetch.
// ---------------------------------------------------------------------------

#define NATOMS 14
#define CTDIM  384
#define TOKENS 65536
#define TOK_PER_BLOCK 256
#define NUM_BLOCKS (TOKENS / TOK_PER_BLOCK)  // 256

// LDS row pitches in dwords (multiples of 4 -> 16B-aligned rows, bank stride 4)
#define P_E1 196   // K=384 -> 192 dw + 4 pad
#define P_D1 116   // K=196 -> pad to 224 -> 112 dw + 4 pad (zeros)
#define P_SM 68    // K=128 -> 64 dw + 4 pad

typedef __attribute__((ext_vector_type(16))) __bf16       bf16x16;
typedef __attribute__((ext_vector_type(2)))  __bf16       bf16x2;
typedef __attribute__((ext_vector_type(8)))  float        f32x8;
typedef __attribute__((ext_vector_type(8)))  unsigned int u32x8;

#define F8Z ((f32x8){0.f,0.f,0.f,0.f,0.f,0.f,0.f,0.f})

__device__ __forceinline__ unsigned int pk2(float a, float b) {
  bf16x2 p; p[0] = (__bf16)a; p[1] = (__bf16)b;
  return __builtin_bit_cast(unsigned int, p);
}
__device__ __forceinline__ f32x8 wmma_bf16(bf16x16 a, bf16x16 b, f32x8 c) {
  return __builtin_amdgcn_wmma_f32_16x16x32_bf16(
      false, a, false, b, (short)0, c, false, false);
}
// Load one bf16 fragment half-row: 2 x ds_load_b128.
__device__ __forceinline__ bf16x16 ldsB(const unsigned int* p) {
  uint4 lo = *reinterpret_cast<const uint4*>(p);
  uint4 hi = *reinterpret_cast<const uint4*>(p + 8);
  u32x8 r = {lo.x, lo.y, lo.z, lo.w, hi.x, hi.y, hi.z, hi.w};
  return __builtin_bit_cast(bf16x16, r);
}
__device__ __forceinline__ bf16x16 cvt16(float4 q0, float4 q1, float4 q2, float4 q3) {
  bf16x16 a;
  a[0]=(__bf16)q0.x; a[1]=(__bf16)q0.y; a[2]=(__bf16)q0.z; a[3]=(__bf16)q0.w;
  a[4]=(__bf16)q1.x; a[5]=(__bf16)q1.y; a[6]=(__bf16)q1.z; a[7]=(__bf16)q1.w;
  a[8]=(__bf16)q2.x; a[9]=(__bf16)q2.y; a[10]=(__bf16)q2.z; a[11]=(__bf16)q2.w;
  a[12]=(__bf16)q3.x; a[13]=(__bf16)q3.y; a[14]=(__bf16)q3.z; a[15]=(__bf16)q3.w;
  return a;
}

__global__ __launch_bounds__(256) void seqhead_kernel(
    const float* __restrict__ A_I,   // [B*N, 384]
    const float* __restrict__ X_L,   // [B*N*14, 3]
    const float* __restrict__ W_d1, const float* __restrict__ b_d1,  // [128,196],[128]
    const float* __restrict__ W_d2, const float* __restrict__ b_d2,  // [64,128],[64]
    const float* __restrict__ W_e1, const float* __restrict__ b_e1,  // [128,384],[128]
    const float* __restrict__ W_e2, const float* __restrict__ b_e2,  // [64,128],[64]
    const float* __restrict__ W_f,  const float* __restrict__ b_f,   // [32,128],[32]
    float* __restrict__ seq_out,     // [B*N, 32]
    int* __restrict__ idx_out)       // [B*N]
{
  __shared__ unsigned int We1_u[128 * P_E1];
  __shared__ unsigned int Wd1_u[128 * P_D1];
  __shared__ unsigned int We2_u[64 * P_SM];
  __shared__ unsigned int Wd2_u[64 * P_SM];
  __shared__ unsigned int Wf_u [32 * P_SM];
  __shared__ unsigned int stage_u[8 * 16 * P_SM];   // per-wave [16 rows][68 dw]
  __shared__ unsigned int fused_u[8 * 16 * P_SM];

  const int tid = threadIdx.x;

  // ---- cooperative weight staging (nested: no div/mod, immediate offsets) --
  {
    // W_e1: 2 threads per row, 49 uint2-pairs each (pairs >= 96 are pad->0)
    int n = tid >> 1, h = tid & 1;
    const float* gs = W_e1 + n * 384;
    unsigned int* ls = We1_u + n * P_E1;
#pragma unroll
    for (int it = 0; it < 49; ++it) {
      int q = h * 49 + it, k = q * 4;
      uint2 d = {0u, 0u};
      if (k < 384) {
        float4 f = *reinterpret_cast<const float4*>(gs + k);
        d.x = pk2(f.x, f.y); d.y = pk2(f.z, f.w);
      }
      *reinterpret_cast<uint2*>(ls + q * 2) = d;
    }
    // W_d1: 2 threads per row, 29 pairs each (k >= 196 are pad->0)
    gs = W_d1 + n * 196;
    ls = Wd1_u + n * P_D1;
#pragma unroll
    for (int it = 0; it < 29; ++it) {
      int q = h * 29 + it, k = q * 4;
      uint2 d = {0u, 0u};
      if (k < 196) {
        float4 f = *reinterpret_cast<const float4*>(gs + k);
        d.x = pk2(f.x, f.y); d.y = pk2(f.z, f.w);
      }
      *reinterpret_cast<uint2*>(ls + q * 2) = d;
    }
  }
  {
    // W_e2/W_d2: 4 threads per row (64 rows), 9 pairs each w/ q<34 guard
    int n = tid >> 2, f4 = tid & 3;
    const float* ge = W_e2 + n * 128;
    const float* gd = W_d2 + n * 128;
    unsigned int* le = We2_u + n * P_SM;
    unsigned int* ld = Wd2_u + n * P_SM;
#pragma unroll
    for (int it = 0; it < 9; ++it) {
      int q = f4 * 9 + it;
      if (q < 34) {
        int k = q * 4;
        uint2 d = {0u, 0u}, d2 = {0u, 0u};
        if (k < 128) {
          float4 f = *reinterpret_cast<const float4*>(ge + k);
          d.x = pk2(f.x, f.y); d.y = pk2(f.z, f.w);
          f = *reinterpret_cast<const float4*>(gd + k);
          d2.x = pk2(f.x, f.y); d2.y = pk2(f.z, f.w);
        }
        *reinterpret_cast<uint2*>(le + q * 2) = d;
        *reinterpret_cast<uint2*>(ld + q * 2) = d2;
      }
    }
  }
  if (tid < 32 * 4) {
    // W_f: 4 threads per row (32 rows), 9 pairs each w/ q<34 guard
    int n = tid >> 2, f4 = tid & 3;
    const float* gs = W_f + n * 128;
    unsigned int* ls = Wf_u + n * P_SM;
#pragma unroll
    for (int it = 0; it < 9; ++it) {
      int q = f4 * 9 + it;
      if (q < 34) {
        int k = q * 4;
        uint2 d = {0u, 0u};
        if (k < 128) {
          float4 f = *reinterpret_cast<const float4*>(gs + k);
          d.x = pk2(f.x, f.y); d.y = pk2(f.z, f.w);
        }
        *reinterpret_cast<uint2*>(ls + q * 2) = d;
      }
    }
  }
  __syncthreads();

  const int lid  = tid & 31;
  const int wave = tid >> 5;
  const int nn   = lid & 15;   // token row (A) / output column (B,C) in tile
  const int g    = lid >> 4;   // lane half
  const int g4   = g * 4;      // dword offset of this half's K quads

  unsigned int* stgu = &stage_u[wave * 16 * P_SM];
  unsigned int* fsdu = &fused_u[wave * 16 * P_SM];
  __bf16*       stgb = (__bf16*)stgu;   // pitch 2*P_SM halves
  __bf16*       fsdb = (__bf16*)fsdu;

  // C-tile writer: lane owns column `col`, rows g*8+r
  auto storeAct = [&](f32x8 acc, int col, const float* bv, __bf16* dst, bool relu) {
    float bias = bv[col];
#pragma unroll
    for (int r = 0; r < 8; ++r) {
      float v = acc[r] + bias;
      if (relu) v = v > 0.f ? v : 0.f;
      dst[(g * 8 + r) * (2 * P_SM) + col] = (__bf16)v;
    }
  };

#pragma unroll 1
  for (int tt = 0; tt < 2; ++tt) {     // two sequential 16-token tiles / wave
    const int t0   = blockIdx.x * TOK_PER_BLOCK + wave * 32 + tt * 16;
    const int tokA = t0 + nn;

    // =============== Phase E1: h_e = relu(A_I @ W_e1^T + b_e1) =============
    {
      bf16x16 afr[12];
      const float* arow = A_I + (size_t)tokA * CTDIM;
#pragma unroll
      for (int ks = 0; ks < 12; ++ks) {
        const float* pk = arow + ks * 32 + g * 8;
        float4 q0 = *reinterpret_cast<const float4*>(pk);
        float4 q1 = *reinterpret_cast<const float4*>(pk + 4);
        float4 q2 = *reinterpret_cast<const float4*>(pk + 16);
        float4 q3 = *reinterpret_cast<const float4*>(pk + 20);
        afr[ks] = cvt16(q0, q1, q2, q3);
      }
      if (tt == 0) {   // prefetch next tile's inputs into cache during compute
        const float* nrow = arow + 16 * CTDIM;
        __builtin_prefetch(nrow, 0, 1);
        __builtin_prefetch(nrow + 128, 0, 1);
        __builtin_prefetch(nrow + 256, 0, 1);
        __builtin_prefetch(X_L + (size_t)(tokA + 16) * NATOMS * 3, 0, 1);
      }
#pragma unroll
      for (int np = 0; np < 4; ++np) {          // output tiles 2np, 2np+1
        f32x8 acc0 = F8Z, acc1 = F8Z;
        const unsigned int* r0 = We1_u + (np * 32 + nn) * P_E1;
        const unsigned int* r1 = r0 + 16 * P_E1;
#pragma unroll
        for (int ks = 0; ks < 12; ++ks) {
          int off = ks * 16 + g4;
          acc0 = wmma_bf16(afr[ks], ldsB(r0 + off), acc0);
          acc1 = wmma_bf16(afr[ks], ldsB(r1 + off), acc1);
        }
        storeAct(acc0, np * 32 + nn,      b_e1, stgb, true);
        storeAct(acc1, np * 32 + 16 + nn, b_e1, stgb, true);
      }
    }
    asm volatile("s_wait_dscnt 0" ::: "memory");

    // =============== Phase E2: embed_out = h_e @ W_e2^T + b_e2 =============
    {
      bf16x16 afs[4];
#pragma unroll
      for (int ks = 0; ks < 4; ++ks)
        afs[ks] = ldsB(stgu + nn * P_SM + ks * 16 + g4);
#pragma unroll
      for (int np = 0; np < 2; ++np) {
        f32x8 acc0 = F8Z, acc1 = F8Z;
        const unsigned int* r0 = We2_u + (np * 32 + nn) * P_SM;
        const unsigned int* r1 = r0 + 16 * P_SM;
#pragma unroll
        for (int ks = 0; ks < 4; ++ks) {
          int off = ks * 16 + g4;
          acc0 = wmma_bf16(afs[ks], ldsB(r0 + off), acc0);
          acc1 = wmma_bf16(afs[ks], ldsB(r1 + off), acc1);
        }
        storeAct(acc0, 64 + np * 32 + nn,      b_e2, fsdb, false);
        storeAct(acc1, 64 + np * 32 + 16 + nn, b_e2, fsdb, false);
      }
    }

    // =============== Phase D1: relu(dist_feats @ W_d1^T + b_d1) ============
    {
      float cx[NATOMS], cy[NATOMS], cz[NATOMS];
      const float* xb = X_L + (size_t)tokA * NATOMS * 3;
#pragma unroll
      for (int a = 0; a < NATOMS; ++a) {
        cx[a] = xb[a * 3 + 0]; cy[a] = xb[a * 3 + 1]; cz[a] = xb[a * 3 + 2];
      }
      // dist feature at compile-time k; canonicalized (min,max) for CSE
      auto dfeat = [&](int k) -> float {
        if (k >= 196) return 0.f;
        int i = k / NATOMS, j = k - (k / NATOMS) * NATOMS;
        int ii = i < j ? i : j, jj = i < j ? j : i;
        float dx = cx[ii] - cx[jj], dy = cy[ii] - cy[jj], dz = cz[ii] - cz[jj];
        return __builtin_sqrtf(dx * dx + dy * dy + dz * dz);  // sqrt(0)==0
      };

      bf16x16 afd[7];
#pragma unroll
      for (int ks = 0; ks < 7; ++ks) {
        bf16x16 a;
        if (g == 0) {   // all k compile-time within each half -> regs only
#pragma unroll
          for (int e = 0; e < 16; ++e) {
            const int k = ks * 32 + (e & 7) + ((e >> 3) << 4);
            a[e] = (__bf16)dfeat(k);
          }
        } else {
#pragma unroll
          for (int e = 0; e < 16; ++e) {
            const int k = ks * 32 + 8 + (e & 7) + ((e >> 3) << 4);
            a[e] = (__bf16)dfeat(k);
          }
        }
        afd[ks] = a;
      }
#pragma unroll
      for (int np = 0; np < 4; ++np) {
        f32x8 acc0 = F8Z, acc1 = F8Z;
        const unsigned int* r0 = Wd1_u + (np * 32 + nn) * P_D1;
        const unsigned int* r1 = r0 + 16 * P_D1;
#pragma unroll
        for (int ks = 0; ks < 7; ++ks) {
          int off = ks * 16 + g4;
          acc0 = wmma_bf16(afd[ks], ldsB(r0 + off), acc0);
          acc1 = wmma_bf16(afd[ks], ldsB(r1 + off), acc1);
        }
        storeAct(acc0, np * 32 + nn,      b_d1, stgb, true);   // overwrite h_e
        storeAct(acc1, np * 32 + 16 + nn, b_d1, stgb, true);
      }
    }
    asm volatile("s_wait_dscnt 0" ::: "memory");

    // =============== Phase D2: dist_out = h_d @ W_d2^T + b_d2 ==============
    {
      bf16x16 afs[4];
#pragma unroll
      for (int ks = 0; ks < 4; ++ks)
        afs[ks] = ldsB(stgu + nn * P_SM + ks * 16 + g4);
#pragma unroll
      for (int np = 0; np < 2; ++np) {
        f32x8 acc0 = F8Z, acc1 = F8Z;
        const unsigned int* r0 = Wd2_u + (np * 32 + nn) * P_SM;
        const unsigned int* r1 = r0 + 16 * P_SM;
#pragma unroll
        for (int ks = 0; ks < 4; ++ks) {
          int off = ks * 16 + g4;
          acc0 = wmma_bf16(afs[ks], ldsB(r0 + off), acc0);
          acc1 = wmma_bf16(afs[ks], ldsB(r1 + off), acc1);
        }
        storeAct(acc0, np * 32 + nn,      b_d2, fsdb, false);  // fused[:,0:64]
        storeAct(acc1, np * 32 + 16 + nn, b_d2, fsdb, false);
      }
    }
    asm volatile("s_wait_dscnt 0" ::: "memory");

    // =============== Phase F: Seq_I = fused @ W_f^T + b_f ==================
    {
      bf16x16 afs[4];
#pragma unroll
      for (int ks = 0; ks < 4; ++ks)
        afs[ks] = ldsB(fsdu + nn * P_SM + ks * 16 + g4);
      f32x8 acc0 = F8Z, acc1 = F8Z;
      const unsigned int* r0 = Wf_u + nn * P_SM;
      const unsigned int* r1 = r0 + 16 * P_SM;
#pragma unroll
      for (int ks = 0; ks < 4; ++ks) {
        int off = ks * 16 + g4;
        acc0 = wmma_bf16(afs[ks], ldsB(r0 + off), acc0);
        acc1 = wmma_bf16(afs[ks], ldsB(r1 + off), acc1);
      }
      // stash Seq_I rows (fp32) into stage buffer (pitch P_SM floats)
      float* sf = (float*)stgu;
      float bias0 = b_f[nn], bias1 = b_f[16 + nn];
#pragma unroll
      for (int r = 0; r < 8; ++r) {
        int m = g * 8 + r;
        sf[m * P_SM + nn]      = acc0[r] + bias0;
        sf[m * P_SM + 16 + nn] = acc1[r] + bias1;
      }
    }
    asm volatile("s_wait_dscnt 0" ::: "memory");

    // =============== epilogue: store Seq_I + first-max argmax ==============
    if (lid < 16) {
      int t = t0 + lid;
      const float* row = (const float*)stgu + lid * P_SM;  // 16B-aligned
      float* orow = seq_out + (size_t)t * 32;
      float best = -3.402823466e+38f;
      int bi = 0;
#pragma unroll
      for (int n4 = 0; n4 < 8; ++n4) {
        float4 v = *reinterpret_cast<const float4*>(row + n4 * 4);
        *reinterpret_cast<float4*>(orow + n4 * 4) = v;
        if (v.x > best) { best = v.x; bi = n4 * 4 + 0; }
        if (v.y > best) { best = v.y; bi = n4 * 4 + 1; }
        if (v.z > best) { best = v.z; bi = n4 * 4 + 2; }
        if (v.w > best) { best = v.w; bi = n4 * 4 + 3; }
      }
      idx_out[t] = bi;
    }
    asm volatile("s_wait_dscnt 0" ::: "memory");  // stage reuse in next tile
  }
}

// ---------------------------------------------------------------------------
extern "C" void kernel_launch(void* const* d_in, const int* in_sizes, int n_in,
                              void* d_out, int out_size, void* d_ws, size_t ws_size,
                              hipStream_t stream) {
  (void)in_sizes; (void)n_in; (void)out_size; (void)d_ws; (void)ws_size;
  // 0 A_I, 1 Q_L (unused), 2 X_L, 3 atom_to_token_map (identity, unused),
  // 4 W_d1, 5 b_d1, 6 W_d2, 7 b_d2, 8 W_e1, 9 b_e1, 10 W_e2, 11 b_e2,
  // 12 W_f, 13 b_f
  const float* A_I  = (const float*)d_in[0];
  const float* X_L  = (const float*)d_in[2];
  const float* W_d1 = (const float*)d_in[4];
  const float* b_d1 = (const float*)d_in[5];
  const float* W_d2 = (const float*)d_in[6];
  const float* b_d2 = (const float*)d_in[7];
  const float* W_e1 = (const float*)d_in[8];
  const float* b_e1 = (const float*)d_in[9];
  const float* W_e2 = (const float*)d_in[10];
  const float* b_e2 = (const float*)d_in[11];
  const float* W_f  = (const float*)d_in[12];
  const float* b_f  = (const float*)d_in[13];

  float* seq = (float*)d_out;                      // [B*N*32] Seq_I
  int*   idx = (int*)(seq + (size_t)TOKENS * 32);  // [B*N] argmax indices

  seqhead_kernel<<<NUM_BLOCKS, 256, 0, stream>>>(
      A_I, X_L, W_d1, b_d1, W_d2, b_d2, W_e1, b_e1, W_e2, b_e2, W_f, b_f,
      seq, idx);
}
